// Attention_35734127902836
// MI455X (gfx1250) — matrix operand order repaired and checked by reading, not compiled, not run on hardware
//
#include <hip/hip_runtime.h>

// ---------------- CDNA5 types ----------------
typedef __attribute__((ext_vector_type(16))) __bf16 v16bf;
typedef __attribute__((ext_vector_type(8)))  __bf16 v8bf;
typedef __attribute__((ext_vector_type(8)))  float  v8f;

static constexpr int B_  = 8;
static constexpr int T1_ = 2048;
static constexpr int T2_ = 512;
static constexpr int C_  = 512;
static constexpr int H_  = 8;
static constexpr int Dh_ = 64;

// ---------------- async global->LDS (CDNA5 GLOBAL_LOAD_ASYNC_TO_LDS_B128) ----
// clang signature (from diagnostic): (v4i addrspace(1)* src, v4i addrspace(3)* dst,
//                                     imm int offset, imm int cpol)
#if __has_builtin(__builtin_amdgcn_global_load_async_to_lds_b128) && \
    __has_builtin(__builtin_amdgcn_s_wait_asynccnt)
#define HAS_ASYNC_LDS 1
typedef int v4i_gcc __attribute__((vector_size(16)));
typedef __attribute__((address_space(1))) v4i_gcc* gas_v4i;
typedef __attribute__((address_space(3))) v4i_gcc* las_v4i;
#endif

__device__ __forceinline__ void async_copy_b128(const void* g, void* l) {
#ifdef HAS_ASYNC_LDS
  __builtin_amdgcn_global_load_async_to_lds_b128((gas_v4i)g, (las_v4i)l, 0, 0);
#else
  *(uint4*)l = *(const uint4*)g;
#endif
}
__device__ __forceinline__ void async_wait0() {
#ifdef HAS_ASYNC_LDS
  __builtin_amdgcn_s_wait_asynccnt(0);
#endif
}

__device__ __forceinline__ v8f v8f_zero() {
  v8f z = {0.f, 0.f, 0.f, 0.f, 0.f, 0.f, 0.f, 0.f};
  return z;
}

__device__ __forceinline__ v8f wmma_bf16(v16bf a, v16bf b, v8f c) {
  // D = A(16x32 bf16) * B(32x16 bf16) + C(16x16 f32) -> v_wmma_f32_16x16x32_bf16
  return __builtin_amdgcn_wmma_f32_16x16x32_bf16(false, a, false, b, (short)0, c,
                                                 false, false);
}

// two 16B LDS/global loads -> one 16-element bf16 fragment register block
__device__ __forceinline__ v16bf ld2x8(const __bf16* p0, const __bf16* p1) {
  v8bf a = *(const v8bf*)p0;
  v8bf b = *(const v8bf*)p1;
  v16bf r;
#pragma unroll
  for (int i = 0; i < 8; ++i) { r[i] = a[i]; r[i + 8] = b[i]; }
  return r;
}

// wave32 butterfly reductions via ds_bpermute
__device__ __forceinline__ float wred_max(float v, int lane) {
#pragma unroll
  for (int m = 16; m >= 1; m >>= 1) {
    int o = __builtin_amdgcn_ds_bpermute((lane ^ m) << 2, __float_as_int(v));
    v = fmaxf(v, __int_as_float(o));
  }
  return v;
}
__device__ __forceinline__ float wred_sum(float v, int lane) {
#pragma unroll
  for (int m = 16; m >= 1; m >>= 1) {
    int o = __builtin_amdgcn_ds_bpermute((lane ^ m) << 2, __float_as_int(v));
    v += __int_as_float(o);
  }
  return v;
}

// ---------------- f32 -> bf16 conversion ----------------
__global__ void cvt_f32_bf16(const float* __restrict__ src, __bf16* __restrict__ dst,
                             int n) {
  for (int i = blockIdx.x * blockDim.x + threadIdx.x; i < n;
       i += gridDim.x * blockDim.x)
    dst[i] = (__bf16)src[i];
}

// ---------------- bf16 GEMM: C[M,N] = A[M,K] * B[K,N] ----------------
// block: 256 threads = 8 waves arranged 4(M) x 2(N); block tile 128x128;
// each wave: 32x64 = 2x4 wmma tiles; K staged 32 at a time in LDS.
// A tile is DMA'd via GLOBAL_LOAD_ASYNC_TO_LDS_B128 when available.
template <bool OUT_BF16>
__global__ void gemm_bf16(const __bf16* __restrict__ A, const __bf16* __restrict__ Bm,
                          void* __restrict__ Cout, int M, int N, int K) {
  __shared__ __align__(16) __bf16 Al[128][32];   // [m][k]
  __shared__ __align__(16) __bf16 Bl[128][32];   // transposed: [n][k]

  const int t = threadIdx.x;
  const int lane = t & 31, wave = t >> 5;
  const int wm = wave & 3, wn = wave >> 2;
  const int hi = lane >> 4, l15 = lane & 15;
  const int rowbase = blockIdx.x * 128, colbase = blockIdx.y * 128;

  v8f acc[2][4];
#pragma unroll
  for (int i = 0; i < 2; ++i)
#pragma unroll
    for (int j = 0; j < 4; ++j) acc[i][j] = v8f_zero();

  const int Arow = t >> 1, Acol = (t & 1) * 16;   // 128 rows x 32 cols, 32B/thread
  const int Bkr = t >> 3, Bcol = (t & 7) * 16;    // 32 rows x 128 cols, 32B/thread

  for (int k0 = 0; k0 < K; k0 += 32) {
    // B tile: global -> registers early (latency overlap across the barrier)
    const __bf16* gb = Bm + (size_t)(k0 + Bkr) * N + (colbase + Bcol);
    __align__(16) __bf16 breg[16];
    *(uint4*)&breg[0] = *(const uint4*)(gb);
    *(uint4*)&breg[8] = *(const uint4*)(gb + 8);
    const __bf16* ga = A + (size_t)(rowbase + Arow) * K + (k0 + Acol);
    if (k0 + 32 < K) {  // global_prefetch_b8 of next tiles
      __builtin_prefetch(ga + 32, 0, 0);
      __builtin_prefetch(gb + (size_t)32 * N, 0, 0);
    }
    __syncthreads();                        // previous iteration done reading LDS
    // A tile: async DMA straight into LDS (2x b128 per thread)
    async_copy_b128(ga, &Al[Arow][Acol]);
    async_copy_b128(ga + 8, &Al[Arow][Acol + 8]);
#pragma unroll
    for (int j = 0; j < 16; ++j) Bl[Bcol + j][Bkr] = breg[j];  // transpose store
    async_wait0();                          // s_wait_asynccnt 0
    __syncthreads();

    v16bf af[2], bfr[4];
#pragma unroll
    for (int mt = 0; mt < 2; ++mt) {
      // A frag: lane = row, elems 0..7 = K (hi?8:0)..+7, elems 8..15 = +16
      const __bf16* p = &Al[wm * 32 + mt * 16 + l15][hi ? 8 : 0];
      af[mt] = ld2x8(p, p + 16);
    }
#pragma unroll
    for (int nt = 0; nt < 4; ++nt) {
      // B frag: lane = col, elems 0..7 = K (hi?8:0)..+7, elems 8..15 = +16
      const __bf16* p = &Bl[wn * 64 + nt * 16 + l15][hi ? 8 : 0];
      bfr[nt] = ld2x8(p, p + 16);
    }
#pragma unroll
    for (int mt = 0; mt < 2; ++mt)
#pragma unroll
      for (int nt = 0; nt < 4; ++nt)
        acc[mt][nt] = wmma_bf16(af[mt], bfr[nt], acc[mt][nt]);
  }

#pragma unroll
  for (int mt = 0; mt < 2; ++mt)
#pragma unroll
    for (int nt = 0; nt < 4; ++nt)
#pragma unroll
      for (int i = 0; i < 8; ++i) {
        // C/D layout: VGPR i -> M = i + 8*hi, N = lane&15
        const int m = rowbase + wm * 32 + mt * 16 + i + hi * 8;
        const int n = colbase + wn * 64 + nt * 16 + l15;
        const float v = acc[mt][nt][i];
        if (OUT_BF16)
          ((__bf16*)Cout)[(size_t)m * N + n] = (__bf16)v;
        else
          ((float*)Cout)[(size_t)m * N + n] = v;
      }
}

// ---------------- fused attention ----------------
// block = (b,h) x 64 queries; 128 threads = 4 waves, one 16-query strip per wave.
// Dynamic LDS: KV 64KB (K[512][64] then Vt[64][512]) | S 128KB f32[64][512]
//            | P 64KB bf16[64][512] | addm 2KB f32[512]  => 258KB (<320KB/WGP)
__global__ void attn_fused(const __bf16* __restrict__ qb, const __bf16* __restrict__ kvb,
                           const unsigned char* __restrict__ mmask,
                           const float* __restrict__ prior,
                           float* __restrict__ score_out, float* __restrict__ prob_out,
                           __bf16* __restrict__ yb) {
  extern __shared__ char smem[];
  __bf16* KV = (__bf16*)smem;
  float* S = (float*)(smem + 65536);
  __bf16* P = (__bf16*)(smem + 65536 + 131072);
  float* addm = (float*)(smem + 65536 + 131072 + 65536);

  const int bh = blockIdx.x;
  const int b = bh >> 3, h = bh & 7;
  const int qbase = blockIdx.y * 64;
  const int t = threadIdx.x, lane = t & 31, wave = t >> 5;
  const int hi = lane >> 4, l15 = lane & 15;
  const float scale = 0.125f;  // Dh^-0.5

  // K tile -> LDS via async DMA (K natural [t2][d] layout == B-operand layout)
  {
    const __bf16* ks = kvb + (size_t)b * T2_ * (2 * C_) + h * Dh_;
    for (int r = t; r < T2_; r += 128) {
      const __bf16* src = ks + (size_t)r * (2 * C_);
      __bf16* dst = KV + r * 64;
#pragma unroll
      for (int j = 0; j < 8; ++j) async_copy_b128(src + 8 * j, dst + 8 * j);
    }
  }

  // Q fragments live in registers for the whole kernel (16x64 per wave)
  v16bf qf[2];
  {
    const int qrow = qbase + wave * 16 + l15;
    const __bf16* qp = qb + ((size_t)(b * T1_ + qrow)) * C_ + h * Dh_;
#pragma unroll
    for (int s = 0; s < 2; ++s) {
      const __bf16* p = qp + s * 32 + hi * 8;
      qf[s] = ld2x8(p, p + 16);
    }
  }

  // additive mask
  for (int i = t; i < T2_; i += 128)
    addm[i] = mmask[b * T2_ + i] ? 0.f : -__builtin_inff();

  async_wait0();   // K tile resident in LDS
  __syncthreads();

  // S = scale * Q K^T + mask; write attn_score and keep in LDS
  for (int nt = 0; nt < 32; ++nt) {
    v8f acc = v8f_zero();
#pragma unroll
    for (int s = 0; s < 2; ++s) {
      const __bf16* p = KV + (nt * 16 + l15) * 64 + s * 32 + hi * 8;
      v16bf kf = ld2x8(p, p + 16);
      acc = wmma_bf16(qf[s], kf, acc);
    }
    const int t2 = nt * 16 + l15;
    const float am = addm[t2];
#pragma unroll
    for (int i = 0; i < 8; ++i) {
      const int m = wave * 16 + i + hi * 8;
      const float sv = acc[i] * scale + am;
      S[m * T2_ + t2] = sv;
      score_out[((size_t)bh * T1_ + qbase + m) * T2_ + t2] = sv;
    }
  }
  __syncthreads();  // all waves done reading K; S complete

  // z = S + log(prior + eps)  (softmax(log_softmax(S)+logp) == softmax(S+logp))
  for (int i4 = t; i4 < 64 * T2_ / 4; i4 += 128) {
    const int flat = i4 * 4, r = flat >> 9, c = flat & 511;
    float4 pv = *(const float4*)(prior + ((size_t)(b * T1_ + qbase + r)) * T2_ + c);
    float4 sv = *(float4*)(S + r * T2_ + c);
    sv.x += __logf(pv.x + 1e-8f);
    sv.y += __logf(pv.y + 1e-8f);
    sv.z += __logf(pv.z + 1e-8f);
    sv.w += __logf(pv.w + 1e-8f);
    *(float4*)(S + r * T2_ + c) = sv;
  }
  // V -> LDS transposed Vt[d][t2] (reuses K buffer; safe after the barrier above)
  {
    const __bf16* vs = kvb + (size_t)b * T2_ * (2 * C_) + C_ + h * Dh_;
    for (int r = t; r < T2_; r += 128) {
      __align__(16) __bf16 vreg[64];
      const uint4* src = (const uint4*)(vs + (size_t)r * (2 * C_));
#pragma unroll
      for (int j = 0; j < 8; ++j) *(uint4*)&vreg[j * 8] = src[j];
#pragma unroll
      for (int d = 0; d < 64; ++d) KV[d * T2_ + r] = vreg[d];
    }
  }
  __syncthreads();

  // softmax per row (wave-parallel, ds_bpermute reductions); write attn_prob
  for (int r = 0; r < 16; ++r) {
    const int row = wave * 16 + r;
    const float* srow = S + row * T2_;
    float mx = -__builtin_inff();
    for (int c = lane; c < T2_; c += 32) mx = fmaxf(mx, srow[c]);
    mx = wred_max(mx, lane);
    float sum = 0.f;
    for (int c = lane; c < T2_; c += 32) sum += __expf(srow[c] - mx);
    sum = wred_sum(sum, lane);
    const float inv = 1.f / sum;
    float* prow = prob_out + ((size_t)bh * T1_ + qbase + row) * T2_;
    for (int c = lane; c < T2_; c += 32) {
      const float p = __expf(srow[c] - mx) * inv;
      prow[c] = p;                 // masked cols: exp(-inf)=0 matches where(mask,..,0)
      P[row * T2_ + c] = (__bf16)p;
    }
  }
  __syncthreads();

  // y = P @ V  (16x512 * 512x64 per wave, K-loop of 16 wmma slices)
  v8f yacc[4];
#pragma unroll
  for (int nt = 0; nt < 4; ++nt) yacc[nt] = v8f_zero();
  for (int ks = 0; ks < 16; ++ks) {
    const __bf16* pp = P + (wave * 16 + l15) * T2_ + ks * 32 + hi * 8;
    v16bf pf = ld2x8(pp, pp + 16);
#pragma unroll
    for (int nt = 0; nt < 4; ++nt) {
      const __bf16* vp = KV + (nt * 16 + l15) * T2_ + ks * 32 + hi * 8;
      v16bf vf = ld2x8(vp, vp + 16);
      yacc[nt] = wmma_bf16(pf, vf, yacc[nt]);
    }
  }
#pragma unroll
  for (int nt = 0; nt < 4; ++nt)
#pragma unroll
    for (int i = 0; i < 8; ++i) {
      const int m = wave * 16 + i + hi * 8;
      const int d = nt * 16 + l15;
      yb[((size_t)(b * T1_ + qbase + m)) * C_ + h * Dh_ + d] = (__bf16)yacc[nt][i];
    }
}

// ---------------- host launcher ----------------
extern "C" void kernel_launch(void* const* d_in, const int* in_sizes, int n_in,
                              void* d_out, int out_size, void* d_ws, size_t ws_size,
                              hipStream_t stream) {
  (void)in_sizes; (void)n_in; (void)out_size; (void)ws_size;
  const float* query = (const float*)d_in[0];
  // d_in[1] = query_mask (unused by reference math)
  const float* memory = (const float*)d_in[2];
  const unsigned char* memory_mask = (const unsigned char*)d_in[3];
  const float* attn_prior = (const float*)d_in[4];
  const float* Wq = (const float*)d_in[5];
  const float* Wkv = (const float*)d_in[6];
  const float* Wo = (const float*)d_in[7];

  // workspace carve (bf16), ~65MB total
  __bf16* w = (__bf16*)d_ws;
  __bf16* qbuf = w;  w += (size_t)B_ * T1_ * C_;        // q projection
  __bf16* kvbuf = w; w += (size_t)B_ * T2_ * 2 * C_;    // kv projection
  __bf16* ybuf = w;  w += (size_t)B_ * T1_ * C_;        // attention output
  __bf16* qcvt = w;  w += (size_t)B_ * T1_ * C_;        // query bf16
  __bf16* mcvt = w;  w += (size_t)B_ * T2_ * C_;        // memory bf16
  __bf16* Wqb = w;   w += (size_t)C_ * C_;
  __bf16* Wkvb = w;  w += (size_t)C_ * 2 * C_;
  __bf16* Wob = w;   w += (size_t)C_ * C_;

  float* out_p = (float*)d_out;
  float* prob_p = out_p + (size_t)B_ * T1_ * C_;
  float* score_p = prob_p + (size_t)B_ * H_ * T1_ * T2_;

  cvt_f32_bf16<<<2048, 256, 0, stream>>>(query, qcvt, B_ * T1_ * C_);
  cvt_f32_bf16<<<1024, 256, 0, stream>>>(memory, mcvt, B_ * T2_ * C_);
  cvt_f32_bf16<<<256, 256, 0, stream>>>(Wq, Wqb, C_ * C_);
  cvt_f32_bf16<<<512, 256, 0, stream>>>(Wkv, Wkvb, C_ * 2 * C_);
  cvt_f32_bf16<<<256, 256, 0, stream>>>(Wo, Wob, C_ * C_);

  dim3 blk(256);
  dim3 gq(B_ * T1_ / 128, C_ / 128);          // 128 x 4
  gemm_bf16<true><<<gq, blk, 0, stream>>>(qcvt, Wqb, qbuf, B_ * T1_, C_, C_);
  dim3 gkv(B_ * T2_ / 128, 2 * C_ / 128);     // 32 x 8
  gemm_bf16<true><<<gkv, blk, 0, stream>>>(mcvt, Wkvb, kvbuf, B_ * T2_, 2 * C_, C_);

  const int smem_bytes = 65536 + 131072 + 65536 + 2048;  // 258KB < 320KB/WGP
  (void)hipFuncSetAttribute((const void*)attn_fused,
                            hipFuncAttributeMaxDynamicSharedMemorySize, smem_bytes);
  dim3 ga(B_ * H_, T1_ / 64);                 // 64 x 32 blocks, 128 threads
  attn_fused<<<ga, dim3(128), smem_bytes, stream>>>(qbuf, kvbuf, memory_mask,
                                                    attn_prior, score_p, prob_p, ybuf);

  gemm_bf16<false><<<gq, blk, 0, stream>>>(ybuf, Wob, (void*)out_p, B_ * T1_, C_, C_);
}
